// Attention2d_50079318671461
// MI455X (gfx1250) — matrix operand order repaired and checked by reading, not compiled, not run on hardware
//
#include <hip/hip_runtime.h>
#include <hip/hip_bf16.h>

typedef __bf16 bf16_t;
typedef __attribute__((ext_vector_type(16))) __bf16 v16bf;
typedef __attribute__((ext_vector_type(8)))  float  v8f;

#define S_DIM   32
#define H_DIM   8
#define R_DIM   384
#define D_MODEL 256
#define HD      32
#define MASK_BIAS (-1.0e9f)

__device__ __forceinline__ v8f wmma_bf16(v16bf a, v16bf b, v8f c) {
  // D = A(16x32 bf16) x B(32x16 bf16) + C(16x16 f32)
  return __builtin_amdgcn_wmma_f32_16x16x32_bf16(
      /*neg_a=*/false, a, /*neg_b=*/false, b,
      /*c_mod=*/(short)0, c, /*reuse_a=*/false, /*reuse_b=*/false);
}

// ---------------------------------------------------------------------------
// Pack fp32 weight matrix W[N][K] (torch Linear layout, y = x W^T) into
// WMMA B-operand tile layout, bf16. Tile (kt, nt): 32 lanes x 16 bf16,
// lane = (khalf<<4)|n16 ; element e -> K = kt*32 + khalf*8 + e + (e>=8 ? 8 : 0), N = nt*16+n16.
// Packed index: ((kt*NT + nt)*32 + lane)*16 + e  -> 32B contiguous per lane.
// ---------------------------------------------------------------------------
__global__ void pack_w_kernel(const float* __restrict__ W, bf16_t* __restrict__ dst,
                              int NT, int KT) {
  int t = blockIdx.x * blockDim.x + threadIdx.x;   // one thread per (tile, lane)
  int total = KT * NT * 32;
  if (t >= total) return;
  int lane = t & 31;
  int tile = t >> 5;
  int nt = tile % NT;
  int kt = tile / NT;
  int n = nt * 16 + (lane & 15);
  int kbase = kt * 32 + (lane >> 4) * 8;
  int K = KT * 32;
  bf16_t* out = dst + (size_t)t * 16;
#pragma unroll
  for (int e = 0; e < 16; ++e) out[e] = (bf16_t)W[(size_t)n * K + kbase + e + (e >= 8 ? 8 : 0)];
}

// ---------------------------------------------------------------------------
// Stage 1: QKV = X @ Wqkv^T + b.  X:[12288,256] fp32, out split into
//   qh,kh : bf16 [S][H][R][32]      (row-major per head)
//   vT    : bf16 [S][H][32][R]      (transposed for PV B-operand loads)
// Block = 256 thr (8 waves), computes 16 rows x 768 cols; wave w owns 6 n-tiles.
// ---------------------------------------------------------------------------
__global__ void qkv_kernel(const float* __restrict__ X, const float* __restrict__ bq,
                           const bf16_t* __restrict__ Wp,
                           bf16_t* __restrict__ qh, bf16_t* __restrict__ kh,
                           bf16_t* __restrict__ vT) {
  const int lane = threadIdx.x & 31;
  const int wave = threadIdx.x >> 5;
  const int half = lane >> 4;
  const int l16  = lane & 15;
  const int m0 = blockIdx.x * 16;

  v8f acc[6] = {};
#pragma unroll 2
  for (int kt = 0; kt < 8; ++kt) {
    const float* arow = X + (size_t)(m0 + l16) * D_MODEL + kt * 32;
    if (kt < 7) __builtin_prefetch(arow + 32, 0, 1);
    v16bf a;
#pragma unroll
    for (int e = 0; e < 8; ++e) a[e]     = (bf16_t)arow[half * 8 + e];
#pragma unroll
    for (int e = 0; e < 8; ++e) a[8 + e] = (bf16_t)arow[16 + half * 8 + e];
#pragma unroll
    for (int t = 0; t < 6; ++t) {
      int nt = wave * 6 + t;
      v16bf b = *(const v16bf*)(Wp + (((size_t)(kt * 48 + nt)) * 32 + lane) * 16);
      acc[t] = wmma_bf16(a, b, acc[t]);
    }
  }
  const int s  = m0 / R_DIM;          // 16-row tile never straddles an s boundary
  const int r0 = m0 % R_DIM;
#pragma unroll
  for (int t = 0; t < 6; ++t) {
    int nt = wave * 6 + t;
    int n = nt * 16 + l16;
    float bias = bq[n];
    int sec = n >> 8;                 // 0=q 1=k 2=v
    int within = n & 255;
    int h = within >> 5, d = within & 31;
#pragma unroll
    for (int r = 0; r < 8; ++r) {
      int rr = r0 + r + 8 * half;
      bf16_t bv = (bf16_t)(acc[t][r] + bias);
      if (sec == 0)      qh[(((size_t)(s * H_DIM + h) * R_DIM) + rr) * HD + d] = bv;
      else if (sec == 1) kh[(((size_t)(s * H_DIM + h) * R_DIM) + rr) * HD + d] = bv;
      else               vT[(((size_t)(s * H_DIM + h) * HD) + d) * R_DIM + rr] = bv;
    }
  }
}

// ---------------------------------------------------------------------------
// Stage 2: per (s,h) flash attention with post-softmax multiplicative bias.
// Block = 256 thr (8 waves); wave w handles q-row tiles {w, w+8, w+16}.
// Online softmax: m,Z in fp32; O accumulated via WMMA with rescaling.
// ---------------------------------------------------------------------------
__global__ void attn_kernel(const bf16_t* __restrict__ qh, const bf16_t* __restrict__ kh,
                            const bf16_t* __restrict__ vT, const int* __restrict__ mask,
                            const float* __restrict__ bias, bf16_t* __restrict__ O) {
  __shared__ bf16_t ldsP[8][16][32];  // per-wave C->A relayout staging (1KB/wave)
  const int lane = threadIdx.x & 31;
  const int wave = threadIdx.x >> 5;
  const int half = lane >> 4;
  const int l16  = lane & 15;
  const int s = blockIdx.x >> 3;
  const int h = blockIdx.x & 7;

  const bf16_t* qb = qh + ((size_t)(s * H_DIM + h) * R_DIM) * HD;
  const bf16_t* kb = kh + ((size_t)(s * H_DIM + h) * R_DIM) * HD;
  const bf16_t* vb = vT + ((size_t)(s * H_DIM + h) * HD) * R_DIM;
  const int*   mrow = mask + s * R_DIM;
  const float* brow = bias + (size_t)h * R_DIM * R_DIM;

  for (int rt = 0; rt < 3; ++rt) {
    const int i0 = (wave + rt * 8) * 16;
    const bf16_t* qrow = qb + (size_t)(i0 + l16) * HD;
    v16bf qa;
#pragma unroll
    for (int e = 0; e < 8; ++e) qa[e]     = qrow[half * 16 + e];
#pragma unroll
    for (int e = 0; e < 8; ++e) qa[8 + e] = qrow[half * 16 + 8 + e];

    float mrun[8], Z[8];
    v8f o0 = {}, o1 = {};
#pragma unroll
    for (int r = 0; r < 8; ++r) { mrun[r] = -3.0e38f; Z[r] = 0.0f; }

    for (int j0 = 0; j0 < R_DIM; j0 += 32) {
      v16bf k0 = *(const v16bf*)(kb + (size_t)(j0 + l16) * HD + half * 16);
      v16bf k1 = *(const v16bf*)(kb + (size_t)(j0 + 16 + l16) * HD + half * 16);
      v8f zc = {};
      v8f s0 = wmma_bf16(qa, k0, zc);     // logits cols j0..j0+15
      v8f s1 = wmma_bf16(qa, k1, zc);     // logits cols j0+16..j0+31
      const int mk0 = mrow[j0 + l16];
      const int mk1 = mrow[j0 + 16 + l16];
#pragma unroll
      for (int r = 0; r < 8; ++r) {
        if (mk0) s0[r] = MASK_BIAS;
        if (mk1) s1[r] = MASK_BIAS;
      }
      float p0[8], p1[8];
#pragma unroll
      for (int r = 0; r < 8; ++r) {
        // row-wise max across the 16 lanes holding this row (stays in half-wave)
        float t = fmaxf(s0[r], s1[r]);
        t = fmaxf(t, __shfl_xor(t, 1));
        t = fmaxf(t, __shfl_xor(t, 2));
        t = fmaxf(t, __shfl_xor(t, 4));
        t = fmaxf(t, __shfl_xor(t, 8));
        float mn = fmaxf(mrun[r], t);
        float sc = __expf(mrun[r] - mn);
        mrun[r] = mn;
        float e0 = __expf(s0[r] - mn);
        float e1 = __expf(s1[r] - mn);
        float rs = e0 + e1;
        rs += __shfl_xor(rs, 1);
        rs += __shfl_xor(rs, 2);
        rs += __shfl_xor(rs, 4);
        rs += __shfl_xor(rs, 8);
        Z[r] = Z[r] * sc + rs;            // denominator from un-biased P
        o0[r] *= sc; o1[r] *= sc;         // rescale running O
        int i = i0 + r + 8 * half;
        p0[r] = e0 * brow[(size_t)i * R_DIM + j0 + l16];        // post-softmax bias
        p1[r] = e1 * brow[(size_t)i * R_DIM + j0 + 16 + l16];
      }
      // C-layout -> A-layout via per-wave LDS (same-wave ds ordering)
#pragma unroll
      for (int r = 0; r < 8; ++r) {
        int row = r + 8 * half;
        ldsP[wave][row][l16]      = (bf16_t)p0[r];
        ldsP[wave][row][16 + l16] = (bf16_t)p1[r];
      }
      v16bf pa;
#pragma unroll
      for (int e = 0; e < 8; ++e) pa[e]     = ldsP[wave][l16][half * 16 + e];
#pragma unroll
      for (int e = 0; e < 8; ++e) pa[8 + e] = ldsP[wave][l16][half * 16 + 8 + e];

      v16bf v0 = *(const v16bf*)(vb + (size_t)(l16) * R_DIM + j0 + half * 16);
      v16bf v1 = *(const v16bf*)(vb + (size_t)(16 + l16) * R_DIM + j0 + half * 16);
      o0 = wmma_bf16(pa, v0, o0);         // O[:,0:16]  += P x V
      o1 = wmma_bf16(pa, v1, o1);         // O[:,16:32] += P x V
    }
#pragma unroll
    for (int r = 0; r < 8; ++r) {
      int i = i0 + r + 8 * half;
      float inv = 1.0f / Z[r];
      size_t base = ((size_t)(s * R_DIM + i)) * D_MODEL + h * HD;
      O[base + l16]      = (bf16_t)(o0[r] * inv);
      O[base + 16 + l16] = (bf16_t)(o1[r] * inv);
    }
  }
}

// ---------------------------------------------------------------------------
// Stage 3: out = Oact @ Wout^T + b_out (fp32 result).
// Block = 256 thr (8 waves), 16 rows x 256 cols; wave owns 2 n-tiles.
// ---------------------------------------------------------------------------
__global__ void outproj_kernel(const bf16_t* __restrict__ A, const bf16_t* __restrict__ Wp,
                               const float* __restrict__ bo, float* __restrict__ out) {
  const int lane = threadIdx.x & 31;
  const int wave = threadIdx.x >> 5;
  const int half = lane >> 4;
  const int l16  = lane & 15;
  const int m0 = blockIdx.x * 16;

  v8f acc[2] = {};
#pragma unroll 2
  for (int kt = 0; kt < 8; ++kt) {
    const bf16_t* arow = A + (size_t)(m0 + l16) * D_MODEL + kt * 32;
    v16bf a;
#pragma unroll
    for (int e = 0; e < 8; ++e) a[e]     = arow[half * 8 + e];
#pragma unroll
    for (int e = 0; e < 8; ++e) a[8 + e] = arow[16 + half * 8 + e];
#pragma unroll
    for (int t = 0; t < 2; ++t) {
      int nt = wave * 2 + t;
      v16bf b = *(const v16bf*)(Wp + (((size_t)(kt * 16 + nt)) * 32 + lane) * 16);
      acc[t] = wmma_bf16(a, b, acc[t]);
    }
  }
#pragma unroll
  for (int t = 0; t < 2; ++t) {
    int n = (wave * 2 + t) * 16 + l16;
    float bias = bo[n];
#pragma unroll
    for (int r = 0; r < 8; ++r)
      out[(size_t)(m0 + r + 8 * half) * D_MODEL + n] = acc[t][r] + bias;
  }
}

// ---------------------------------------------------------------------------
extern "C" void kernel_launch(void* const* d_in, const int* in_sizes, int n_in,
                              void* d_out, int out_size, void* d_ws, size_t ws_size,
                              hipStream_t stream) {
  (void)in_sizes; (void)n_in; (void)out_size; (void)ws_size;
  const float* pair_act = (const float*)d_in[0];
  const int*   amask    = (const int*)d_in[1];
  const float* bias     = (const float*)d_in[2];
  const float* W_qkv    = (const float*)d_in[3];
  const float* b_qkv    = (const float*)d_in[4];
  const float* W_out    = (const float*)d_in[5];
  const float* b_out    = (const float*)d_in[6];

  const size_t per = (size_t)S_DIM * H_DIM * R_DIM * HD;  // 3,145,728 bf16 elems
  bf16_t* qh    = (bf16_t*)d_ws;
  bf16_t* kh    = qh + per;
  bf16_t* vT    = kh + per;
  bf16_t* Oact  = vT + per;                 // [S*R][256] == same element count
  bf16_t* WqkvP = Oact + per;               // 768*256
  bf16_t* WoutP = WqkvP + (size_t)768 * 256;  // 256*256   (total ~25.7 MB)

  pack_w_kernel<<<(8 * 48 * 32 + 255) / 256, 256, 0, stream>>>(W_qkv, WqkvP, 48, 8);
  pack_w_kernel<<<(8 * 16 * 32 + 255) / 256, 256, 0, stream>>>(W_out, WoutP, 16, 8);
  qkv_kernel   <<<768, 256, 0, stream>>>(pair_act, b_qkv, WqkvP, qh, kh, vT);
  attn_kernel  <<<256, 256, 0, stream>>>(qh, kh, vT, amask, bias, Oact);
  outproj_kernel<<<768, 256, 0, stream>>>(Oact, WoutP, b_out, (float*)d_out);
}
